// Attention_86217173500445
// MI455X (gfx1250) — compile-verified, matrix operand order checked by reading, not run on hardware
//
#include <hip/hip_runtime.h>
#include <hip/hip_bf16.h>

// MI455X / gfx1250: wave32, WMMA 16x16x32 bf16 with f32 accumulation.
// Phases: (1) weight convert+transpose to bf16, (2) QKV GEMMs (f32 A -> bf16),
// (3) flash-style 4-stream attention (V staged transposed in LDS),
// (4) output projection GEMM -> f32 d_out.

typedef __attribute__((ext_vector_type(16))) __bf16 v16bf;
typedef __attribute__((ext_vector_type(8)))  float  v8f;

union FragBF {
    v16bf v;
    uint4 q[2];
    unsigned short u[16];
};

__device__ __forceinline__ unsigned short f2bf(float f) {
    unsigned u = __float_as_uint(f);
    u += 0x7FFFu + ((u >> 16) & 1u);   // round-to-nearest-even
    return (unsigned short)(u >> 16);
}

#define SEQ   197
#define CDIM  768
#define NHEAD 12
#define HD    64
#define MROWS 6304          // 32 * 197
#define ATTN_SCALE 0.125f   // 64^-0.5

// ---- gfx1250 async global->LDS copy (ASYNCcnt path), guarded ---------------
#if defined(__AMDGCN__) && __has_builtin(__builtin_amdgcn_global_load_async_to_lds_b128)
#define HAVE_ASYNC_LDS 1
#else
#define HAVE_ASYNC_LDS 0
#endif

#if HAVE_ASYNC_LDS
typedef int vint4 __attribute__((vector_size(16)));
typedef __attribute__((address_space(1))) vint4* as1_vint4p;
typedef __attribute__((address_space(3))) vint4* as3_vint4p;

__device__ __forceinline__ void async_copy_b128(const void* g, void* l) {
    __builtin_amdgcn_global_load_async_to_lds_b128(
        (as1_vint4p)(unsigned long long)g,
        (as3_vint4p)(unsigned)(unsigned long long)l,
        0, 0);
}
__device__ __forceinline__ void async_wait0() {
#if __has_builtin(__builtin_amdgcn_s_wait_asynccnt)
    __builtin_amdgcn_s_wait_asynccnt(0);
#else
    asm volatile("s_wait_asynccnt 0x0" ::: "memory");
#endif
}
#endif

// ---------------------------------------------------------------------------
// Weight convert + transpose: W [K=768][N=768] f32 -> Wt [N][K] bf16
// ---------------------------------------------------------------------------
__global__ __launch_bounds__(256)
void conv_w_transpose(const float* __restrict__ W, unsigned short* __restrict__ Wt) {
    int idx = blockIdx.x * 256 + threadIdx.x;
    if (idx < CDIM * CDIM) {
        int k = idx / CDIM, n = idx % CDIM;
        Wt[(size_t)n * CDIM + k] = f2bf(W[idx]);
    }
}

// ---------------------------------------------------------------------------
// Tiled GEMM: C[M,768] = A[M,768] @ Wt^T + bias
//   block tile 128x128, 8 waves (2x4), each wave 64x32 -> 8 wmma / K-step
// ---------------------------------------------------------------------------
template<bool A_F32, bool OUT_F32>
__global__ __launch_bounds__(256)
void gemm_bias_kernel(const void* __restrict__ Aptr,
                      const unsigned short* __restrict__ Wt,   // bf16 [N][K]
                      const float* __restrict__ bias,
                      void* __restrict__ Cptr,
                      int M, int Mtiles)
{
    __shared__ __align__(16) unsigned short As[128][40]; // 80B rows, 16B aligned
    __shared__ __align__(16) unsigned short Bs[128][40];

    const int t    = threadIdx.x;
    const int lane = t & 31;
    const int wv   = t >> 5;
    const int wm   = wv & 1;        // 0..1 : 64-row half
    const int wn   = wv >> 1;       // 0..3 : 32-col quarter
    const int lh   = lane >> 4;     // lane half (K-striping)
    const int ln   = lane & 15;     // M/N within 16

    const int mt = blockIdx.x % Mtiles;
    const int nt = blockIdx.x / Mtiles;
    const int m0 = mt * 128;
    const int n0 = nt * 128;

    v8f acc[4][2];
#pragma unroll
    for (int i = 0; i < 4; ++i)
#pragma unroll
        for (int j = 0; j < 2; ++j) acc[i][j] = {};

    const int srow  = t >> 1;           // 0..127 staged row
    const int scol  = (t & 1) * 16;     // 0 or 16
    const int agrow = min(m0 + srow, M - 1);
    const int bgrow = n0 + srow;

    for (int ks = 0; ks < 24; ++ks) {
        const int k0 = ks * 32;

        // ---- stage A tile ----
        if (A_F32) {
            const float* src = (const float*)Aptr + (size_t)agrow * CDIM + k0 + scol;
#pragma unroll
            for (int q4 = 0; q4 < 4; ++q4) {
                float4 f = ((const float4*)src)[q4];
                As[srow][scol + q4 * 4 + 0] = f2bf(f.x);
                As[srow][scol + q4 * 4 + 1] = f2bf(f.y);
                As[srow][scol + q4 * 4 + 2] = f2bf(f.z);
                As[srow][scol + q4 * 4 + 3] = f2bf(f.w);
            }
        } else {
            const uint4* src = (const uint4*)((const unsigned short*)Aptr +
                                              (size_t)agrow * CDIM + k0 + scol);
#if HAVE_ASYNC_LDS
            async_copy_b128(src + 0, &As[srow][scol]);
            async_copy_b128(src + 1, &As[srow][scol + 8]);
#else
            *(uint4*)&As[srow][scol]     = src[0];
            *(uint4*)&As[srow][scol + 8] = src[1];
#endif
        }
        // ---- stage B tile (Wt rows = output cols) ----
        {
            const uint4* src = (const uint4*)(Wt + (size_t)bgrow * CDIM + k0 + scol);
#if HAVE_ASYNC_LDS
            async_copy_b128(src + 0, &Bs[srow][scol]);
            async_copy_b128(src + 1, &Bs[srow][scol + 8]);
#else
            *(uint4*)&Bs[srow][scol]     = src[0];
            *(uint4*)&Bs[srow][scol + 8] = src[1];
#endif
        }
        if (ks + 1 < 24) { // prefetch next K-step of B -> global_prefetch_b8
            __builtin_prefetch((const void*)(Wt + (size_t)bgrow * CDIM + k0 + 32 + scol), 0, 1);
        }
#if HAVE_ASYNC_LDS
        async_wait0();
#endif
        __syncthreads();

        FragBF af[4], bfr[2];
#pragma unroll
        for (int i = 0; i < 4; ++i) {
            const int mr = wm * 64 + i * 16 + ln;
            af[i].q[0] = *(const uint4*)&As[mr][lh * 8];
            af[i].q[1] = *(const uint4*)&As[mr][lh * 8 + 16];
        }
#pragma unroll
        for (int j = 0; j < 2; ++j) {
            const int nr = wn * 32 + j * 16 + ln;
            bfr[j].q[0] = *(const uint4*)&Bs[nr][lh * 8];
            bfr[j].q[1] = *(const uint4*)&Bs[nr][lh * 8 + 16];
        }
#pragma unroll
        for (int i = 0; i < 4; ++i)
#pragma unroll
            for (int j = 0; j < 2; ++j)
                acc[i][j] = __builtin_amdgcn_wmma_f32_16x16x32_bf16(
                    false, af[i].v, false, bfr[j].v, (short)0, acc[i][j], false, false);
        __syncthreads();
    }

    // Epilogue: C/D layout -> VGPR r holds rows r (lanes 0-15) / r+8 (lanes 16-31)
#pragma unroll
    for (int j = 0; j < 2; ++j) {
        const int n  = n0 + wn * 32 + j * 16 + ln;
        const float bv = bias[n];
#pragma unroll
        for (int i = 0; i < 4; ++i) {
#pragma unroll
            for (int r = 0; r < 8; ++r) {
                const int m = m0 + wm * 64 + i * 16 + lh * 8 + r;
                if (m < M) {
                    const float v = acc[i][j][r] + bv;
                    if (OUT_F32)
                        ((float*)Cptr)[(size_t)m * CDIM + n] = v;
                    else
                        ((unsigned short*)Cptr)[(size_t)m * CDIM + n] = f2bf(v);
                }
            }
        }
    }
}

// ---------------------------------------------------------------------------
// Flash-style attention: one wave per (stream, b, h, 16-query tile)
//   4 waves / block; 4*384*13 = 19968 units -> 4992 blocks exactly
// ---------------------------------------------------------------------------
__global__ __launch_bounds__(128)
void attn_kernel(const unsigned short* __restrict__ q_b, const unsigned short* __restrict__ k_b,
                 const unsigned short* __restrict__ v_b, const unsigned short* __restrict__ q_t,
                 const unsigned short* __restrict__ k_t, const unsigned short* __restrict__ v_t,
                 unsigned short* __restrict__ outp)
{
    __shared__ __align__(16) float          s_sc[4][16][32];
    __shared__ __align__(16) unsigned short s_p[4][16][32];
    __shared__ __align__(16) unsigned short s_vt[4][64][40]; // V^T [d][key], 80B rows
    __shared__ float s_m[4][16];
    __shared__ float s_l[4][16];
    __shared__ float s_c[4][16];

    const int lane = threadIdx.x & 31;
    const int w    = threadIdx.x >> 5;
    const int lh   = lane >> 4;
    const int ln   = lane & 15;

    const int unit   = blockIdx.x * 4 + w;
    const int stream = unit / (384 * 13);
    int rem          = unit % (384 * 13);
    const int bh     = rem / 13;
    const int qtile  = rem % 13;
    const int b      = bh / NHEAD;
    const int h      = bh % NHEAD;

    const unsigned short* Q = (stream & 1) ? q_t : q_b;          // 0:b 1:t 2:b 3:t
    const bool kvT = (stream == 1) || (stream == 2);             // 0:b 1:t 2:t 3:b
    const unsigned short* K = kvT ? k_t : k_b;
    const unsigned short* V = kvT ? v_t : v_b;

    const int rowbase = b * SEQ;
    const int hd      = h * HD;

    // Q A-fragments for d = [0,32) and [32,64)
    FragBF qf[2];
    {
        const int qm = min(qtile * 16 + ln, SEQ - 1);
        const unsigned short* qrow = Q + (size_t)(rowbase + qm) * CDIM + hd;
#pragma unroll
        for (int dh = 0; dh < 2; ++dh) {
            const unsigned short* p = qrow + dh * 32 + lh * 8;
            qf[dh].q[0] = *(const uint4*)p;
            qf[dh].q[1] = *(const uint4*)(p + 16);
        }
    }

    v8f oacc[4];
#pragma unroll
    for (int dt = 0; dt < 4; ++dt) oacc[dt] = {};

    if (lane < 16) { s_m[w][lane] = -1e30f; s_l[w][lane] = 0.0f; }
    __syncthreads();

    for (int kb0 = 0; kb0 < SEQ; kb0 += 32) {
        // ---- stage V^T tile into LDS first: its global loads + ds stores are
        //      independent of the score WMMAs and overlap with them ----
        {
            const int kp   = ln * 2;          // key pair within tile (even)
            const int d0   = lh * 32;         // d half
            const int key0 = min(kb0 + kp,     SEQ - 1);
            const int key1 = min(kb0 + kp + 1, SEQ - 1);
            const unsigned short* v0p = V + (size_t)(rowbase + key0) * CDIM + hd + d0;
            const unsigned short* v1p = V + (size_t)(rowbase + key1) * CDIM + hd + d0;
            union { uint4 q[4]; unsigned short u[32]; } r0, r1;
#pragma unroll
            for (int q4 = 0; q4 < 4; ++q4) {
                r0.q[q4] = ((const uint4*)v0p)[q4];
                r1.q[q4] = ((const uint4*)v1p)[q4];
            }
#pragma unroll
            for (int d = 0; d < 32; ++d) {
                *(unsigned int*)&s_vt[w][d0 + d][kp] =
                    (unsigned)r0.u[d] | ((unsigned)r1.u[d] << 16);
            }
        }

        // ---- scores: S[16 x 32] = Q (16x64) @ K^T (64x32) ----
        // load ALL four K B-fragments before any WMMA so the 8 b128 loads
        // issue as one clause and drain while the matrix pipe works
        FragBF kf[2][2];
#pragma unroll
        for (int kt2 = 0; kt2 < 2; ++kt2) {
            const int key = min(kb0 + kt2 * 16 + ln, SEQ - 1);
            const unsigned short* krow = K + (size_t)(rowbase + key) * CDIM + hd;
#pragma unroll
            for (int dh = 0; dh < 2; ++dh) {
                const unsigned short* p = krow + dh * 32 + lh * 8;
                kf[kt2][dh].q[0] = *(const uint4*)p;
                kf[kt2][dh].q[1] = *(const uint4*)(p + 16);
            }
        }
        v8f sa0 = {}, sa1 = {};
#pragma unroll
        for (int dh = 0; dh < 2; ++dh)
            sa0 = __builtin_amdgcn_wmma_f32_16x16x32_bf16(
                false, qf[dh].v, false, kf[0][dh].v, (short)0, sa0, false, false);
#pragma unroll
        for (int dh = 0; dh < 2; ++dh)
            sa1 = __builtin_amdgcn_wmma_f32_16x16x32_bf16(
                false, qf[dh].v, false, kf[1][dh].v, (short)0, sa1, false, false);

#pragma unroll
        for (int r = 0; r < 8; ++r) {
            const int row = lh * 8 + r;
            s_sc[w][row][ln]      = sa0[r] * ATTN_SCALE;
            s_sc[w][row][16 + ln] = sa1[r] * ATTN_SCALE;
        }
        __syncthreads();

        // ---- online softmax: one lane per query row ----
        if (lane < 16) {
            const int row = lane;
            const float mold = s_m[w][row];
            float mx = mold;
            float sc[32];
#pragma unroll
            for (int c = 0; c < 32; ++c) {
                float sv = s_sc[w][row][c];
                sc[c] = (kb0 + c < SEQ) ? sv : -1e30f;
                mx = fmaxf(mx, sc[c]);
            }
            const float corr = __expf(mold - mx);
            float sum = 0.0f;
#pragma unroll
            for (int c = 0; c < 32; ++c) {
                const float p = (kb0 + c < SEQ) ? __expf(sc[c] - mx) : 0.0f;
                s_p[w][row][c] = f2bf(p);
                sum += p;
            }
            s_m[w][row] = mx;
            s_l[w][row] = s_l[w][row] * corr + sum;
            s_c[w][row] = corr;
        }
        __syncthreads();

        // ---- rescale O and accumulate P @ V (all fragments from LDS, b128) ----
        float cf[8];
#pragma unroll
        for (int r = 0; r < 8; ++r) cf[r] = s_c[w][lh * 8 + r];
#pragma unroll
        for (int dt = 0; dt < 4; ++dt)
#pragma unroll
            for (int r = 0; r < 8; ++r) oacc[dt][r] *= cf[r];

        FragBF pf;
        pf.q[0] = *(const uint4*)&s_p[w][ln][lh * 8];
        pf.q[1] = *(const uint4*)&s_p[w][ln][lh * 8 + 16];

#pragma unroll
        for (int dt = 0; dt < 4; ++dt) {
            FragBF vf;
            vf.q[0] = *(const uint4*)&s_vt[w][dt * 16 + ln][lh * 8];
            vf.q[1] = *(const uint4*)&s_vt[w][dt * 16 + ln][lh * 8 + 16];
            oacc[dt] = __builtin_amdgcn_wmma_f32_16x16x32_bf16(
                false, pf.v, false, vf.v, (short)0, oacc[dt], false, false);
        }
        __syncthreads();
    }

    if (lane < 16) s_c[w][lane] = 1.0f / s_l[w][lane];
    __syncthreads();

    float inv[8];
#pragma unroll
    for (int r = 0; r < 8; ++r) inv[r] = s_c[w][lh * 8 + r];
#pragma unroll
    for (int dt = 0; dt < 4; ++dt)
#pragma unroll
        for (int r = 0; r < 8; ++r) {
            const int srow = qtile * 16 + lh * 8 + r;
            if (srow < SEQ)
                outp[((size_t)stream * MROWS + rowbase + srow) * CDIM + hd + dt * 16 + ln] =
                    f2bf(oacc[dt][r] * inv[r]);
        }
}

// ---------------------------------------------------------------------------
extern "C" void kernel_launch(void* const* d_in, const int* in_sizes, int n_in,
                              void* d_out, int out_size, void* d_ws, size_t ws_size,
                              hipStream_t stream)
{
    const float* xb = (const float*)d_in[0];
    const float* xt = (const float*)d_in[1];
    const float* Wq = (const float*)d_in[2];
    const float* bq = (const float*)d_in[3];
    const float* Wk = (const float*)d_in[4];
    const float* bk = (const float*)d_in[5];
    const float* Wv = (const float*)d_in[6];
    const float* bv = (const float*)d_in[7];
    const float* Wp = (const float*)d_in[8];
    const float* bp = (const float*)d_in[9];
    float* out = (float*)d_out;

    char* ws = (char*)d_ws;
    size_t off = 0;
    auto carve = [&](size_t bytes) -> void* {
        void* p = ws + off;
        off += (bytes + 255) & ~(size_t)255;
        return p;
    };

    const size_t wbytes = (size_t)CDIM * CDIM * 2;
    const size_t mbytes = (size_t)MROWS * CDIM * 2;
    unsigned short* wtq  = (unsigned short*)carve(wbytes);
    unsigned short* wtk  = (unsigned short*)carve(wbytes);
    unsigned short* wtv  = (unsigned short*)carve(wbytes);
    unsigned short* wtp  = (unsigned short*)carve(wbytes);
    unsigned short* qb16 = (unsigned short*)carve(mbytes);
    unsigned short* kb16 = (unsigned short*)carve(mbytes);
    unsigned short* vb16 = (unsigned short*)carve(mbytes);
    unsigned short* qt16 = (unsigned short*)carve(mbytes);
    unsigned short* kt16 = (unsigned short*)carve(mbytes);
    unsigned short* vt16 = (unsigned short*)carve(mbytes);
    unsigned short* attn = (unsigned short*)carve(4 * mbytes);

    // 1) weights -> bf16, transposed
    const int wblk = (CDIM * CDIM + 255) / 256;
    conv_w_transpose<<<wblk, 256, 0, stream>>>(Wq, wtq);
    conv_w_transpose<<<wblk, 256, 0, stream>>>(Wk, wtk);
    conv_w_transpose<<<wblk, 256, 0, stream>>>(Wv, wtv);
    conv_w_transpose<<<wblk, 256, 0, stream>>>(Wp, wtp);

    // 2) QKV projections: M=6304 (50 M-tiles), N=768 (6 N-tiles) -> 300 blocks
    gemm_bias_kernel<true, false><<<300, 256, 0, stream>>>(xb, wtq, bq, qb16, MROWS, 50);
    gemm_bias_kernel<true, false><<<300, 256, 0, stream>>>(xb, wtk, bk, kb16, MROWS, 50);
    gemm_bias_kernel<true, false><<<300, 256, 0, stream>>>(xb, wtv, bv, vb16, MROWS, 50);
    gemm_bias_kernel<true, false><<<300, 256, 0, stream>>>(xt, wtq, bq, qt16, MROWS, 50);
    gemm_bias_kernel<true, false><<<300, 256, 0, stream>>>(xt, wtk, bk, kt16, MROWS, 50);
    gemm_bias_kernel<true, false><<<300, 256, 0, stream>>>(xt, wtv, bv, vt16, MROWS, 50);

    // 3) 4-stream attention: 19968 wave-units / 4 waves per block = 4992 blocks
    attn_kernel<<<4992, 128, 0, stream>>>(qb16, kb16, vb16, qt16, kt16, vt16, attn);

    // 4) output projection to f32 d_out: M=25216 (197 M-tiles) -> 1182 blocks
    gemm_bias_kernel<false, true><<<1182, 256, 0, stream>>>(attn, wtp, bp, out, 4 * MROWS, 197);
}
